// Head_798863917774
// MI455X (gfx1250) — matrix operand order repaired and checked by reading, not compiled, not run on hardware
//
#include <hip/hip_runtime.h>
#include <hip/hip_bf16.h>

// Problem constants (match reference)
#define BATCH 1024
#define T 128
#define C 192
#define H 192

// Padded LDS row strides (elements). Rows 16B-aligned (b128 loads) AND
// bank-perfect: 200*2B=400B=100dw -> bank(36m+c)%64 spacing 4; 136*2B=272B=
// 68dw -> bank(4m+c)%64 spacing 4. 16 lanes x 4 consecutive banks tile all
// 64 banks exactly: conflict-free ds_load_b128.
#define CP 200
#define HP 200
#define TP 136

typedef __bf16 bf16;
typedef __attribute__((ext_vector_type(16))) __bf16 v16bf;
typedef __attribute__((ext_vector_type(8)))  float  v8f;
typedef __attribute__((ext_vector_type(4)))  float  v4f;
typedef __attribute__((ext_vector_type(4)))  unsigned int v4u;

struct U32x8 { v4u lo, hi; };   // 32B, for bit_cast into v16bf

// LDS layout (byte offsets into dynamic smem); all region bases 16B-aligned.
#define XS_OFF  0
#define WT_OFF  (XS_OFF + T * CP * 2)        //  51200
#define QS_OFF  (WT_OFF + H * CP * 2)        // 128000
#define KS_OFF  (QS_OFF + T * HP * 2)        // 179200
#define VST_OFF (KS_OFF + T * HP * 2)        // 230400
#define LDS_BYTES (VST_OFF + H * TP * 2)     // 282624  (< 320KB/WGP)
#define PS_OFF  0                            // P aliases xs (dead after QKV)

// ---- WMMA fragment loaders (contiguous k=16h+s, same order for A and B) ----------------------
// A (16x32 bf16): lane half h, row m=lane&15; VGPR0..3 K=k0+16h+{0..7},
// VGPR4..7 K=k0+16h+8+{0..7}. Source row-major [M][K]; two ds_load_b128.
__device__ __forceinline__ v16bf ldsA(const bf16* base, int stride, int m0,
                                      int k0, int lane) {
  const int hf = (lane >> 4) & 1, m = lane & 15;
  const bf16* p = base + (m0 + m) * stride + k0 + hf * 16;  // 16B-aligned
  U32x8 u;
  u.lo = *(const v4u*)p;          // K = k0+16h .. +7
  u.hi = *(const v4u*)(p + 8);    // K = k0+16h+8 .. +7
  return __builtin_bit_cast(v16bf, u);
}

// B (32x16 bf16): lane half h -> K base k0+16h, col n=lane&15; VGPRv holds
// K = base+2v, base+2v+1. Source stored [N][K]; two ds_load_b128.
__device__ __forceinline__ v16bf ldsB(const bf16* base, int stride, int n0,
                                      int k0, int lane) {
  const int hf = (lane >> 4) & 1, n = lane & 15;
  const bf16* p = base + (n0 + n) * stride + k0 + hf * 16;  // 16B-aligned
  U32x8 u;
  u.lo = *(const v4u*)p;
  u.hi = *(const v4u*)(p + 8);
  return __builtin_bit_cast(v16bf, u);
}

__device__ __forceinline__ v8f wmma_bf16(v16bf a, v16bf b, v8f c) {
  return __builtin_amdgcn_wmma_f32_16x16x32_bf16(false, a, false, b, (short)0,
                                                 c, false, false);
}

// Transposed D-tile store: lane's 8 values are 8 consecutive columns of one
// [N][M] row -> pack to 8 bf16 and emit a single ds_store_b128.
__device__ __forceinline__ void storeD_T(bf16* dst, int dstStride, int m0,
                                         int n0, v8f acc, int lane) {
  const int hf = (lane >> 4) & 1, n = lane & 15;
  union { bf16 e[8]; v4u u; } pk;
#pragma unroll
  for (int r = 0; r < 8; ++r) pk.e[r] = (bf16)acc[r];
  *(v4u*)(dst + (n0 + n) * dstStride + m0 + 8 * hf) = pk.u;   // 16B-aligned
}

// Row-major D-tile store (scatter b16; layout dictated by WMMA C-layout).
__device__ __forceinline__ void storeD(bf16* dst, int dstStride, int m0,
                                       int n0, v8f acc, int lane) {
  const int hf = (lane >> 4) & 1, n = lane & 15;
#pragma unroll
  for (int r = 0; r < 8; ++r)
    dst[(m0 + r + 8 * hf) * dstStride + n0 + n] = (bf16)acc[r];
}

// One wave computes its 16 rows of (xs @ W), dual accumulator chains for ILP.
template <bool TRANSPOSED>
__device__ __forceinline__ void proj_store(const bf16* xs, const bf16* wt,
                                           bf16* dst, int dstStride, int w,
                                           int lane) {
  const int m0 = w * 16;
  v16bf afr[C / 32];
#pragma unroll
  for (int kt = 0; kt < C / 32; ++kt) afr[kt] = ldsA(xs, CP, m0, kt * 32, lane);
#pragma unroll
  for (int nt = 0; nt < H / 16; nt += 2) {
    v8f acc0 = {}, acc1 = {};
#pragma unroll
    for (int kt = 0; kt < C / 32; ++kt) {
      const v16bf b0 = ldsB(wt, CP, nt * 16, kt * 32, lane);
      const v16bf b1 = ldsB(wt, CP, nt * 16 + 16, kt * 32, lane);
      acc0 = wmma_bf16(afr[kt], b0, acc0);
      acc1 = wmma_bf16(afr[kt], b1, acc1);
    }
    if (TRANSPOSED) {
      storeD_T(dst, dstStride, m0, nt * 16, acc0, lane);
      storeD_T(dst, dstStride, m0, nt * 16 + 16, acc1, lane);
    } else {
      storeD(dst, dstStride, m0, nt * 16, acc0, lane);
      storeD(dst, dstStride, m0, nt * 16 + 16, acc1, lane);
    }
  }
}

// Stage a (C x H) f32 weight matrix -> transposed bf16 LDS [H][CP].
__device__ __forceinline__ void stage_w(const float* __restrict__ Wsrc,
                                        bf16* wt, int tid) {
  for (int j = tid; j < C * H / 4; j += 256) {
    const v4f d = ((const v4f*)Wsrc)[j];           // global_load_b128
    const int c = j / (H / 4), h4 = (j % (H / 4)) * 4;
#pragma unroll
    for (int k = 0; k < 4; ++k) wt[(h4 + k) * CP + c] = (bf16)d[k];
  }
}

__global__ __launch_bounds__(256) void head_attn_fused(
    const float* __restrict__ x, const float* __restrict__ Wq,
    const float* __restrict__ Wk, const float* __restrict__ Wv,
    float* __restrict__ out) {
  extern __shared__ char smem[];
  bf16* xs  = (bf16*)(smem + XS_OFF);
  bf16* wt  = (bf16*)(smem + WT_OFF);
  bf16* qs  = (bf16*)(smem + QS_OFF);
  bf16* ks  = (bf16*)(smem + KS_OFF);
  bf16* vst = (bf16*)(smem + VST_OFF);
  bf16* ps  = (bf16*)(smem + PS_OFF);

  const int tid  = threadIdx.x;
  const int lane = tid & 31;
  const int w    = tid >> 5;           // wave id 0..7, owns rows 16w..16w+15
  const int b    = blockIdx.x;
  const float* xb = x + (size_t)b * (T * C);

  // Stage x -> bf16 LDS: float4 loads, packed 4xbf16 (b64) stores.
  for (int j = tid; j < T * C / 4; j += 256) {
    const v4f d = ((const v4f*)xb)[j];
    const int t = j / (C / 4), c4 = (j % (C / 4)) * 4;
    union { bf16 e[4]; unsigned long long u; } pk;
#pragma unroll
    for (int k = 0; k < 4; ++k) pk.e[k] = (bf16)d[k];
    *(unsigned long long*)(xs + t * CP + c4) = pk.u;   // 8B-aligned
  }
  __syncthreads();

  // ---- Projections: Q, K row-major; V transposed ----
  stage_w(Wq, wt, tid);
  __syncthreads();
  proj_store<false>(xs, wt, qs, HP, w, lane);
  __syncthreads();

  stage_w(Wk, wt, tid);
  __syncthreads();
  proj_store<false>(xs, wt, ks, HP, w, lane);
  __syncthreads();

  stage_w(Wv, wt, tid);
  __syncthreads();
  proj_store<true>(xs, wt, vst, TP, w, lane);
  __syncthreads();   // after this: qs/ks/vst valid; xs/wt dead -> ps aliases xs

  // ---- S = Q K^T (causal tiles only), softmax in registers ----
  const int m0 = w * 16;
  const int hf = (lane >> 4) & 1, n = lane & 15;
  const float scale = 0.07216878364870322f;   // C^-0.5 (NOT head^-0.5)

  v16bf qfr[H / 32];
#pragma unroll
  for (int kt = 0; kt < H / 32; ++kt) qfr[kt] = ldsA(qs, HP, m0, kt * 32, lane);

  v8f s[T / 16];
#pragma unroll
  for (int t0 = 0; t0 < T / 16; ++t0) {
    v8f acc = {};
    if (t0 <= w) {
#pragma unroll
      for (int kt = 0; kt < H / 32; ++kt)
        acc = wmma_bf16(qfr[kt], ldsB(ks, HP, t0 * 16, kt * 32, lane), acc);
    }
    s[t0] = acc;
  }

#pragma unroll
  for (int r = 0; r < 8; ++r) {
    const int row = m0 + r + 8 * hf;
    float mx = -3.0e38f;
#pragma unroll
    for (int t0 = 0; t0 < T / 16; ++t0) {
      const int col = t0 * 16 + n;
      float v = s[t0][r] * scale;
      v = (t0 <= w && col <= row) ? v : -3.0e38f;
      s[t0][r] = v;
      mx = fmaxf(mx, v);
    }
    for (int off = 1; off < 16; off <<= 1)
      mx = fmaxf(mx, __shfl_xor(mx, off, 32));   // reduce over N (half-wave)
    float sum = 0.0f;
#pragma unroll
    for (int t0 = 0; t0 < T / 16; ++t0) {
      const float e = (s[t0][r] > -1.0e38f) ? __expf(s[t0][r] - mx) : 0.0f;
      s[t0][r] = e;
      sum += e;
    }
    for (int off = 1; off < 16; off <<= 1) sum += __shfl_xor(sum, off, 32);
    const float inv = 1.0f / sum;
#pragma unroll
    for (int t0 = 0; t0 < T / 16; ++t0) s[t0][r] *= inv;
  }

  // P -> per-wave LDS strip (zeros in masked region => full-tile K loop safe)
  bf16* psw = ps + w * 16 * TP;
#pragma unroll
  for (int r = 0; r < 8; ++r) {
    const int Ml = r + 8 * hf;
#pragma unroll
    for (int t0 = 0; t0 < T / 16; ++t0)
      psw[Ml * TP + t0 * 16 + n] = (bf16)s[t0][r];
  }

  // ---- O = P V (triangular K loop, dual chains), store f32 to HBM ----
  const int nsteps = (w >> 1) + 1;              // ceil(16(w+1)/32)
  v16bf pfr[T / 32];
#pragma unroll
  for (int kt = 0; kt < T / 32; ++kt)
    if (kt < nsteps) pfr[kt] = ldsA(psw, TP, 0, kt * 32, lane);

  float* ob = out + (size_t)b * (T * H);
#pragma unroll
  for (int nt = 0; nt < H / 16; nt += 2) {
    v8f acc0 = {}, acc1 = {};
#pragma unroll
    for (int kt = 0; kt < T / 32; ++kt)
      if (kt < nsteps) {
        const v16bf b0 = ldsB(vst, TP, nt * 16, kt * 32, lane);
        const v16bf b1 = ldsB(vst, TP, nt * 16 + 16, kt * 32, lane);
        acc0 = wmma_bf16(pfr[kt], b0, acc0);
        acc1 = wmma_bf16(pfr[kt], b1, acc1);
      }
#pragma unroll
    for (int r = 0; r < 8; ++r) {
      const int row = (m0 + r + 8 * hf) * H;
      ob[row + nt * 16 + n]      = acc0[r];
      ob[row + nt * 16 + 16 + n] = acc1[r];
    }
  }
}

extern "C" void kernel_launch(void* const* d_in, const int* in_sizes, int n_in,
                              void* d_out, int out_size, void* d_ws,
                              size_t ws_size, hipStream_t stream) {
  const float* x  = (const float*)d_in[0];
  const float* Wq = (const float*)d_in[1];
  const float* Wk = (const float*)d_in[2];
  const float* Wv = (const float*)d_in[3];
  float* out = (float*)d_out;
  (void)in_sizes; (void)n_in; (void)out_size; (void)d_ws; (void)ws_size;

  head_attn_fused<<<dim3(BATCH), dim3(256), LDS_BYTES, stream>>>(x, Wq, Wk, Wv,
                                                                 out);
}
